// DeepSeekMoE_6571299963294
// MI455X (gfx1250) — matrix operand order repaired and checked by reading, not compiled, not run on hardware
//
#include <hip/hip_runtime.h>
#include <hip/hip_bf16.h>
#include <math.h>

// ---------------------------------------------------------------------------
// Shapes (fixed by the reference): B=4, S=1024, IN=512, D=4096, F=16384,
// H=32 heads, dh=128, E=8 experts, OUT=256.  TOK = B*S = 4096 tokens.
// ---------------------------------------------------------------------------
#define TOK   4096
#define DIN   512
#define DMODEL 4096
#define DFF   16384
#define DOUT  256
#define NHEAD 32
#define DHEAD 128
#define SEQ   1024

typedef __bf16 bf16_t;
typedef __attribute__((ext_vector_type(16))) __bf16 bf16x16;
typedef __attribute__((ext_vector_type(8)))  float  floatx8;
typedef __attribute__((ext_vector_type(4)))  unsigned int uint4v;

union FragU { bf16x16 v; uint4v u[2]; };

__device__ __forceinline__ floatx8 wmma_bf16(const FragU& a, const FragU& b, floatx8 c) {
    // D = A(16x32 bf16) * B(32x16 bf16) + C(16x16 f32)
    return __builtin_amdgcn_wmma_f32_16x16x32_bf16(false, a.v, false, b.v,
                                                   (short)0, c, false, false);
}

// CDNA5 async global->LDS copy (16B per lane), tracked with ASYNCcnt.
// Inline asm keeps it portable across the two toolchains' builtin arities.
__device__ __forceinline__ void async_cp16(bf16_t* lds, const bf16_t* gptr) {
    unsigned int loff = (unsigned int)(unsigned long long)lds;  // LDS byte addr
    asm volatile("global_load_async_to_lds_b128 %0, %1, off"
                 :: "v"(loff), "v"(gptr) : "memory");
}
__device__ __forceinline__ void wait_async0() {
    asm volatile("s_wait_asynccnt 0" ::: "memory");
}

// ---------------------------------------------------------------------------
// fp32 -> bf16 elementwise cast (for activations x)
// ---------------------------------------------------------------------------
__global__ void cast_bf16_kernel(const float* __restrict__ src,
                                 bf16_t* __restrict__ dst, int n) {
    int i = blockIdx.x * 256 + threadIdx.x;
    if (i < n) dst[i] = (bf16_t)src[i];
}

// ---------------------------------------------------------------------------
// Tiled transpose + cast: src[R][C] fp32 -> dst[C][R] bf16.
// Weight matrices become [N][K] so GEMM B-fragments are contiguous in K.
// ---------------------------------------------------------------------------
__global__ void tcast_kernel(const float* __restrict__ src,
                             bf16_t* __restrict__ dst, int R, int C) {
    __shared__ float tile[32][33];
    int c0 = blockIdx.x * 32, r0 = blockIdx.y * 32;
    int x = threadIdx.x, y = threadIdx.y;  // block (32, 8)
    for (int i = 0; i < 4; i++)
        tile[y + i * 8][x] = src[(size_t)(r0 + y + i * 8) * C + c0 + x];
    __syncthreads();
    for (int i = 0; i < 4; i++)
        dst[(size_t)(c0 + y + i * 8) * R + r0 + x] = (bf16_t)tile[x][y + i * 8];
}

// ---------------------------------------------------------------------------
// bf16 GEMM: C[M,N] = A[M,K] @ W[K,N], with W stored transposed WT[N][K].
// Tile 128(M) x 64(N) per 256-thread block (8 waves), K-step 32.
// Double-buffered LDS fed by GLOBAL_LOAD_ASYNC_TO_LDS_B128 (ASYNCcnt),
// one barrier per K-step; B fragments preloaded so WMMAs issue back-to-back.
// MODE 0: write bf16      MODE 1: write fp32 + bias    MODE 2: fp32 +=
// ---------------------------------------------------------------------------
template <int MODE>
__global__ __launch_bounds__(256) void gemm_kernel(
        const bf16_t* __restrict__ A, const bf16_t* __restrict__ WT,
        const float* __restrict__ bias, bf16_t* __restrict__ outb,
        float* __restrict__ outf, int M, int N, int K) {
    __shared__ bf16_t As[2][128 * 40];  // padded row stride 40 bf16 (80B)
    __shared__ bf16_t Bs[2][64 * 40];
    const int t = threadIdx.x, lane = t & 31, w = t >> 5;
    const int h = lane >> 4, mr = lane & 15;
    const int n0 = blockIdx.x * 64, m0 = blockIdx.y * 128;
    const int srow = t >> 2, skc = (t & 3) * 8;   // staging coords

    floatx8 acc[4];
#pragma unroll
    for (int i = 0; i < 4; i++)
#pragma unroll
        for (int j = 0; j < 8; j++) acc[i][j] = 0.f;

    // issue async stage of first K-tile
    {
        async_cp16(&As[0][srow * 40 + skc],        &A[(size_t)(m0 + srow) * K + skc]);
        async_cp16(&As[0][(srow + 64) * 40 + skc], &A[(size_t)(m0 + srow + 64) * K + skc]);
        async_cp16(&Bs[0][srow * 40 + skc],        &WT[(size_t)(n0 + srow) * K + skc]);
    }
    wait_async0();
    __syncthreads();

    int p = 0;
    for (int k0 = 0; k0 < K; k0 += 32) {
        if (k0 + 32 < K) {  // prefetch next tile into the other buffer
            int kn = k0 + 32;
            async_cp16(&As[p ^ 1][srow * 40 + skc],        &A[(size_t)(m0 + srow) * K + kn + skc]);
            async_cp16(&As[p ^ 1][(srow + 64) * 40 + skc], &A[(size_t)(m0 + srow + 64) * K + kn + skc]);
            async_cp16(&Bs[p ^ 1][srow * 40 + skc],        &WT[(size_t)(n0 + srow) * K + kn + skc]);
        }
        FragU fa;  // A 16x32: lane m holds K runs at h*8 and 16+h*8
        fa.u[0] = *(const uint4v*)&As[p][(w * 16 + mr) * 40 + h * 8];
        fa.u[1] = *(const uint4v*)&As[p][(w * 16 + mr) * 40 + 16 + h * 8];
        FragU fbs[4];  // B 32x16: lane n holds K runs at h*8 and 16+h*8
#pragma unroll
        for (int nt = 0; nt < 4; nt++) {
            fbs[nt].u[0] = *(const uint4v*)&Bs[p][(nt * 16 + mr) * 40 + h * 8];
            fbs[nt].u[1] = *(const uint4v*)&Bs[p][(nt * 16 + mr) * 40 + 16 + h * 8];
        }
#pragma unroll
        for (int nt = 0; nt < 4; nt++) acc[nt] = wmma_bf16(fa, fbs[nt], acc[nt]);
        wait_async0();
        __syncthreads();
        p ^= 1;
    }
    // C layout: reg r <-> row r+8h, col = lane&15
#pragma unroll
    for (int nt = 0; nt < 4; nt++) {
#pragma unroll
        for (int r = 0; r < 8; r++) {
            int row = m0 + w * 16 + r + 8 * h;
            int col = n0 + nt * 16 + mr;
            size_t idx = (size_t)row * N + col;
            float v = acc[nt][r];
            if (MODE == 0)      outb[idx] = (bf16_t)v;
            else if (MODE == 1) outf[idx] = v + bias[col];
            else                outf[idx] += v;
        }
    }
}

// ---------------------------------------------------------------------------
// Fused gated MLP up-projection: out = silu(A@Wg) * (A@Wu)  (bf16 out)
// Same async double-buffered structure, two B streams, 8 back-to-back WMMAs.
// ---------------------------------------------------------------------------
__global__ __launch_bounds__(256) void gemm_gated_kernel(
        const bf16_t* __restrict__ A, const bf16_t* __restrict__ WgT,
        const bf16_t* __restrict__ WuT, bf16_t* __restrict__ out,
        int M, int N, int K) {
    __shared__ bf16_t As[2][128 * 40];
    __shared__ bf16_t Bg[2][64 * 40];
    __shared__ bf16_t Bu[2][64 * 40];
    const int t = threadIdx.x, lane = t & 31, w = t >> 5;
    const int h = lane >> 4, mr = lane & 15;
    const int n0 = blockIdx.x * 64, m0 = blockIdx.y * 128;
    const int srow = t >> 2, skc = (t & 3) * 8;

    floatx8 accg[4], accu[4];
#pragma unroll
    for (int i = 0; i < 4; i++)
#pragma unroll
        for (int j = 0; j < 8; j++) { accg[i][j] = 0.f; accu[i][j] = 0.f; }

    {
        async_cp16(&As[0][srow * 40 + skc],        &A[(size_t)(m0 + srow) * K + skc]);
        async_cp16(&As[0][(srow + 64) * 40 + skc], &A[(size_t)(m0 + srow + 64) * K + skc]);
        async_cp16(&Bg[0][srow * 40 + skc],        &WgT[(size_t)(n0 + srow) * K + skc]);
        async_cp16(&Bu[0][srow * 40 + skc],        &WuT[(size_t)(n0 + srow) * K + skc]);
    }
    wait_async0();
    __syncthreads();

    int p = 0;
    for (int k0 = 0; k0 < K; k0 += 32) {
        if (k0 + 32 < K) {
            int kn = k0 + 32;
            async_cp16(&As[p ^ 1][srow * 40 + skc],        &A[(size_t)(m0 + srow) * K + kn + skc]);
            async_cp16(&As[p ^ 1][(srow + 64) * 40 + skc], &A[(size_t)(m0 + srow + 64) * K + kn + skc]);
            async_cp16(&Bg[p ^ 1][srow * 40 + skc],        &WgT[(size_t)(n0 + srow) * K + kn + skc]);
            async_cp16(&Bu[p ^ 1][srow * 40 + skc],        &WuT[(size_t)(n0 + srow) * K + kn + skc]);
        }
        FragU fa;
        fa.u[0] = *(const uint4v*)&As[p][(w * 16 + mr) * 40 + h * 8];
        fa.u[1] = *(const uint4v*)&As[p][(w * 16 + mr) * 40 + 16 + h * 8];
        FragU fg[4], fu[4];
#pragma unroll
        for (int nt = 0; nt < 4; nt++) {
            fg[nt].u[0] = *(const uint4v*)&Bg[p][(nt * 16 + mr) * 40 + h * 8];
            fg[nt].u[1] = *(const uint4v*)&Bg[p][(nt * 16 + mr) * 40 + 16 + h * 8];
            fu[nt].u[0] = *(const uint4v*)&Bu[p][(nt * 16 + mr) * 40 + h * 8];
            fu[nt].u[1] = *(const uint4v*)&Bu[p][(nt * 16 + mr) * 40 + 16 + h * 8];
        }
#pragma unroll
        for (int nt = 0; nt < 4; nt++) {
            accg[nt] = wmma_bf16(fa, fg[nt], accg[nt]);
            accu[nt] = wmma_bf16(fa, fu[nt], accu[nt]);
        }
        wait_async0();
        __syncthreads();
        p ^= 1;
    }
#pragma unroll
    for (int nt = 0; nt < 4; nt++) {
#pragma unroll
        for (int r = 0; r < 8; r++) {
            int row = m0 + w * 16 + r + 8 * h;
            int col = n0 + nt * 16 + mr;
            float g = accg[nt][r], u = accu[nt][r];
            float val = (g / (1.f + __expf(-g))) * u;  // silu(g) * u
            out[(size_t)row * N + col] = (bf16_t)val;
        }
    }
}

// ---------------------------------------------------------------------------
// RMSNorm: fp32 h row -> bf16 out row
// ---------------------------------------------------------------------------
__global__ __launch_bounds__(256) void rmsnorm_kernel(
        const float* __restrict__ h, const float* __restrict__ wln,
        bf16_t* __restrict__ out) {
    __shared__ float red[256];
    int row = blockIdx.x, t = threadIdx.x;
    const float* hr = h + (size_t)row * DMODEL;
    float s = 0.f;
    for (int d = t; d < DMODEL; d += 256) { float v = hr[d]; s += v * v; }
    red[t] = s; __syncthreads();
    for (int off = 128; off > 0; off >>= 1) {
        if (t < off) red[t] += red[t + off];
        __syncthreads();
    }
    float rstd = rsqrtf(red[0] / (float)DMODEL + 1e-6f);
    bf16_t* orow = out + (size_t)row * DMODEL;
    for (int d = t; d < DMODEL; d += 256) orow[d] = (bf16_t)(hr[d] * rstd * wln[d]);
}

// ---------------------------------------------------------------------------
// Flash-style causal attention, one (batch, head, 128-q-row block) per block.
// Wave w owns 16 q rows; streams 32-key tiles; WMMA for QK^T and PV.
// ---------------------------------------------------------------------------
__global__ __launch_bounds__(256) void attn_kernel(
        const bf16_t* __restrict__ Q, const bf16_t* __restrict__ Kb,
        const bf16_t* __restrict__ Vb, bf16_t* __restrict__ O) {
    __shared__ bf16_t Kt[32 * 136];       // K tile [key][dh], padded
    __shared__ bf16_t Vt[128 * 40];       // V tile transposed [dh][key]
    __shared__ bf16_t Pw[8 * 16 * 40];    // per-wave P staging [16][32], padded
    const int t = threadIdx.x, lane = t & 31, w = t >> 5;
    const int h = lane >> 4, mr = lane & 15;
    const int q0 = blockIdx.x * 128, hd = blockIdx.y, b = blockIdx.z;
    const float scale = 0.08838834764831845f;  // 1/sqrt(128)

    // Q fragments (4 x 32-channel chunks), held in registers for all key tiles
    const bf16_t* qrow = Q + (size_t)(b * SEQ + q0 + w * 16 + mr) * DMODEL + hd * DHEAD;
    FragU qf[4];
#pragma unroll
    for (int c = 0; c < 4; c++) {
        qf[c].u[0] = *(const uint4v*)&qrow[c * 32 + h * 8];
        qf[c].u[1] = *(const uint4v*)&qrow[c * 32 + 16 + h * 8];
    }

    float Mi[8], Li[8];
    floatx8 ao[8];
#pragma unroll
    for (int r = 0; r < 8; r++) { Mi[r] = -3.0e38f; Li[r] = 0.f; }
#pragma unroll
    for (int cg = 0; cg < 8; cg++)
#pragma unroll
        for (int r = 0; r < 8; r++) ao[cg][r] = 0.f;

    const int ktiles = (q0 + 128) / 32;   // causal upper bound
    for (int kt = 0; kt < ktiles; kt++) {
        // stage K (row-major, async) and V (transposed) tiles
        for (int j = 0; j < 2; j++) {
            int idx = t + j * 256;
            int key = idx >> 4, cc = (idx & 15) * 8;
            const bf16_t* ks = Kb + (size_t)(b * SEQ + kt * 32 + key) * DMODEL + hd * DHEAD + cc;
            async_cp16(&Kt[key * 136 + cc], ks);
            const bf16_t* vs = Vb + (size_t)(b * SEQ + kt * 32 + key) * DMODEL + hd * DHEAD + cc;
#pragma unroll
            for (int jj = 0; jj < 8; jj++) Vt[(cc + jj) * 40 + key] = vs[jj];
        }
        wait_async0();
        __syncthreads();

        // scores = Q @ K^T  (two 16-key columns, dh contracted in 4 chunks)
        floatx8 sacc[2];
#pragma unroll
        for (int kc = 0; kc < 2; kc++)
#pragma unroll
            for (int r = 0; r < 8; r++) sacc[kc][r] = 0.f;
#pragma unroll
        for (int kc = 0; kc < 2; kc++)
#pragma unroll
            for (int c = 0; c < 4; c++) {
                FragU fb;
                fb.u[0] = *(const uint4v*)&Kt[(kc * 16 + mr) * 136 + c * 32 + h * 8];
                fb.u[1] = *(const uint4v*)&Kt[(kc * 16 + mr) * 136 + c * 32 + 16 + h * 8];
                sacc[kc] = wmma_bf16(qf[c], fb, sacc[kc]);
            }

        // online softmax update (row reductions across 16-lane halves)
        float p0v[8], p1v[8];
#pragma unroll
        for (int r = 0; r < 8; r++) {
            int qidx = q0 + w * 16 + r + 8 * h;
            float s0 = sacc[0][r] * scale;
            float s1 = sacc[1][r] * scale;
            if (kt * 32 + mr > qidx)      s0 = -3.0e38f;
            if (kt * 32 + 16 + mr > qidx) s1 = -3.0e38f;
            float rm = fmaxf(s0, s1);
            for (int m = 1; m < 16; m <<= 1) rm = fmaxf(rm, __shfl_xor(rm, m, 32));
            float newM = fmaxf(Mi[r], rm);
            float f = __expf(Mi[r] - newM);
            float p0 = __expf(s0 - newM), p1 = __expf(s1 - newM);
            float rs = p0 + p1;
            for (int m = 1; m < 16; m <<= 1) rs += __shfl_xor(rs, m, 32);
            Li[r] = Li[r] * f + rs;
            Mi[r] = newM;
#pragma unroll
            for (int cg = 0; cg < 8; cg++) ao[cg][r] *= f;
            p0v[r] = p0; p1v[r] = p1;
        }

        // C-layout -> A-fragment layout via wave-private LDS staging
        bf16_t* pw = &Pw[w * 16 * 40];
#pragma unroll
        for (int r = 0; r < 8; r++) {
            pw[(r + 8 * h) * 40 + mr]      = (bf16_t)p0v[r];
            pw[(r + 8 * h) * 40 + 16 + mr] = (bf16_t)p1v[r];
        }
        asm volatile("s_wait_dscnt 0" ::: "memory");
        FragU pf;
        pf.u[0] = *(const uint4v*)&pw[mr * 40 + h * 8];
        pf.u[1] = *(const uint4v*)&pw[mr * 40 + 16 + h * 8];

        // ao += P(16x32) @ V(32 x 128)
#pragma unroll
        for (int cg = 0; cg < 8; cg++) {
            FragU fb;
            fb.u[0] = *(const uint4v*)&Vt[(cg * 16 + mr) * 40 + h * 8];
            fb.u[1] = *(const uint4v*)&Vt[(cg * 16 + mr) * 40 + 16 + h * 8];
            ao[cg] = wmma_bf16(pf, fb, ao[cg]);
        }
        __syncthreads();
    }

    // normalize and store attn output (bf16)
#pragma unroll
    for (int r = 0; r < 8; r++) {
        float inv = 1.f / Li[r];
        bf16_t* orow = O + (size_t)(b * SEQ + q0 + w * 16 + r + 8 * h) * DMODEL + hd * DHEAD;
#pragma unroll
        for (int cg = 0; cg < 8; cg++) orow[cg * 16 + mr] = (bf16_t)(ao[cg][r] * inv);
    }
}

// ---------------------------------------------------------------------------
// Router: logits = h@Wr + br; top-2 on logits (== top-2 on softmax probs);
// gather h[t, idx] (idx in [0,8) indexes FEATURES per the reference) -> mean.
// ---------------------------------------------------------------------------
__global__ __launch_bounds__(256) void router_kernel(
        const float* __restrict__ h, const float* __restrict__ Wr,
        const float* __restrict__ br, float* __restrict__ sbuf) {
    __shared__ float sl[256 * 8];
    int row = blockIdx.x, t = threadIdx.x;
    const float* hr = h + (size_t)row * DMODEL;
    float part[8];
    for (int e = 0; e < 8; e++) part[e] = 0.f;
    for (int d = t; d < DMODEL; d += 256) {
        float hv = hr[d];
        for (int e = 0; e < 8; e++) part[e] += hv * Wr[d * 8 + e];
    }
    for (int e = 0; e < 8; e++) sl[t * 8 + e] = part[e];
    __syncthreads();
    for (int off = 128; off > 0; off >>= 1) {
        if (t < off)
            for (int e = 0; e < 8; e++) sl[t * 8 + e] += sl[(t + off) * 8 + e];
        __syncthreads();
    }
    if (t == 0) {
        float lg[8];
        for (int e = 0; e < 8; e++) lg[e] = sl[e] + br[e];
        int i0 = 0;
        for (int e = 1; e < 8; e++) if (lg[e] > lg[i0]) i0 = e;
        int i1 = -1;
        for (int e = 0; e < 8; e++)
            if (e != i0 && (i1 < 0 || lg[e] > lg[i1])) i1 = e;
        sbuf[row] = 0.5f * (hr[i0] + hr[i1]);
    }
}

// wsum[o] = sum_d Wout[d][o]   (agg @ Wout collapses to s[t] * wsum[o])
__global__ __launch_bounds__(256) void wsum_kernel(
        const float* __restrict__ Wout, float* __restrict__ wsum) {
    __shared__ float red[256];
    int o = blockIdx.x, t = threadIdx.x;
    float s = 0.f;
    for (int d = t; d < DMODEL; d += 256) s += Wout[(size_t)d * DOUT + o];
    red[t] = s; __syncthreads();
    for (int off = 128; off > 0; off >>= 1) {
        if (t < off) red[t] += red[t + off];
        __syncthreads();
    }
    if (t == 0) wsum[o] = red[0];
}

__global__ void final_kernel(const float* __restrict__ sbuf,
                             const float* __restrict__ wsum,
                             const float* __restrict__ bout,
                             float* __restrict__ out) {
    int tk = blockIdx.x, o = threadIdx.x;
    out[(size_t)tk * DOUT + o] = sbuf[tk] * wsum[o] + bout[o];
}

// ---------------------------------------------------------------------------
extern "C" void kernel_launch(void* const* d_in, const int* in_sizes, int n_in,
                              void* d_out, int out_size, void* d_ws, size_t ws_size,
                              hipStream_t stream) {
    const float* x     = (const float*)d_in[0];
    const float* Wi    = (const float*)d_in[1];
    const float* bi    = (const float*)d_in[2];
    const float* ln1_w = (const float*)d_in[3];
    const float* Wq    = (const float*)d_in[4];
    const float* Wk    = (const float*)d_in[5];
    const float* Wv    = (const float*)d_in[6];
    const float* Wo    = (const float*)d_in[7];
    const float* ln2_w = (const float*)d_in[8];
    const float* Wg    = (const float*)d_in[9];
    const float* Wu    = (const float*)d_in[10];
    const float* Wd    = (const float*)d_in[11];
    const float* Wr    = (const float*)d_in[12];
    const float* br    = (const float*)d_in[13];
    const float* Wout  = (const float*)d_in[14];
    const float* bout  = (const float*)d_in[15];
    float* out = (float*)d_out;

    // workspace layout (256B aligned slabs)
    char* base = (char*)d_ws;
    size_t off = 0;
    auto alloc = [&](size_t bytes) {
        size_t o = off;
        off = (off + bytes + 255) & ~(size_t)255;
        return (void*)(base + o);
    };
    bf16_t* WiT = (bf16_t*)alloc((size_t)DMODEL * DIN * 2);     // [4096][512]
    bf16_t* WqT = (bf16_t*)alloc((size_t)DMODEL * DMODEL * 2);
    bf16_t* WkT = (bf16_t*)alloc((size_t)DMODEL * DMODEL * 2);
    bf16_t* WvT = (bf16_t*)alloc((size_t)DMODEL * DMODEL * 2);
    bf16_t* WoT = (bf16_t*)alloc((size_t)DMODEL * DMODEL * 2);
    bf16_t* WgT = (bf16_t*)alloc((size_t)DFF * DMODEL * 2);     // [16384][4096]
    bf16_t* WuT = (bf16_t*)alloc((size_t)DFF * DMODEL * 2);
    bf16_t* WdT = (bf16_t*)alloc((size_t)DMODEL * DFF * 2);     // [4096][16384]
    bf16_t* xb  = (bf16_t*)alloc((size_t)TOK * DIN * 2);
    float*  hbuf= (float*) alloc((size_t)TOK * DMODEL * 4);
    bf16_t* ab  = (bf16_t*)alloc((size_t)TOK * DMODEL * 2);     // rmsnorm out (reused)
    bf16_t* qb  = (bf16_t*)alloc((size_t)TOK * DMODEL * 2);
    bf16_t* kb  = (bf16_t*)alloc((size_t)TOK * DMODEL * 2);
    bf16_t* vb  = (bf16_t*)alloc((size_t)TOK * DMODEL * 2);
    bf16_t* atb = (bf16_t*)alloc((size_t)TOK * DMODEL * 2);
    bf16_t* gub = (bf16_t*)alloc((size_t)TOK * DFF * 2);
    float*  sbuf= (float*) alloc((size_t)TOK * 4);
    float*  wsum= (float*) alloc((size_t)DOUT * 4);
    (void)ws_size; (void)in_sizes; (void)n_in; (void)out_size;

    dim3 tb(32, 8);
    // weight transpose+cast (bf16 copies live in 192MB L2 across GEMM M-tiles)
    tcast_kernel<<<dim3(DMODEL / 32, DIN / 32), tb, 0, stream>>>(Wi, WiT, DIN, DMODEL);
    tcast_kernel<<<dim3(DMODEL / 32, DMODEL / 32), tb, 0, stream>>>(Wq, WqT, DMODEL, DMODEL);
    tcast_kernel<<<dim3(DMODEL / 32, DMODEL / 32), tb, 0, stream>>>(Wk, WkT, DMODEL, DMODEL);
    tcast_kernel<<<dim3(DMODEL / 32, DMODEL / 32), tb, 0, stream>>>(Wv, WvT, DMODEL, DMODEL);
    tcast_kernel<<<dim3(DMODEL / 32, DMODEL / 32), tb, 0, stream>>>(Wo, WoT, DMODEL, DMODEL);
    tcast_kernel<<<dim3(DFF / 32, DMODEL / 32), tb, 0, stream>>>(Wg, WgT, DMODEL, DFF);
    tcast_kernel<<<dim3(DFF / 32, DMODEL / 32), tb, 0, stream>>>(Wu, WuT, DMODEL, DFF);
    tcast_kernel<<<dim3(DMODEL / 32, DFF / 32), tb, 0, stream>>>(Wd, WdT, DFF, DMODEL);
    cast_bf16_kernel<<<(TOK * DIN) / 256, 256, 0, stream>>>(x, xb, TOK * DIN);

    // h = x @ Wi + bi
    gemm_kernel<1><<<dim3(DMODEL / 64, TOK / 128), 256, 0, stream>>>(
        xb, WiT, bi, nullptr, hbuf, TOK, DMODEL, DIN);
    // a = rmsnorm(h, ln1)
    rmsnorm_kernel<<<TOK, 256, 0, stream>>>(hbuf, ln1_w, ab);
    // q,k,v
    gemm_kernel<0><<<dim3(DMODEL / 64, TOK / 128), 256, 0, stream>>>(
        ab, WqT, nullptr, qb, nullptr, TOK, DMODEL, DMODEL);
    gemm_kernel<0><<<dim3(DMODEL / 64, TOK / 128), 256, 0, stream>>>(
        ab, WkT, nullptr, kb, nullptr, TOK, DMODEL, DMODEL);
    gemm_kernel<0><<<dim3(DMODEL / 64, TOK / 128), 256, 0, stream>>>(
        ab, WvT, nullptr, vb, nullptr, TOK, DMODEL, DMODEL);
    // causal flash attention
    attn_kernel<<<dim3(SEQ / 128, NHEAD, TOK / SEQ), 256, 0, stream>>>(qb, kb, vb, atb);
    // h += attn @ Wo
    gemm_kernel<2><<<dim3(DMODEL / 64, TOK / 128), 256, 0, stream>>>(
        atb, WoT, nullptr, nullptr, hbuf, TOK, DMODEL, DMODEL);
    // m = rmsnorm(h, ln2)
    rmsnorm_kernel<<<TOK, 256, 0, stream>>>(hbuf, ln2_w, ab);
    // gu = silu(m@Wg) * (m@Wu)
    gemm_gated_kernel<<<dim3(DFF / 64, TOK / 128), 256, 0, stream>>>(
        ab, WgT, WuT, gub, TOK, DFF, DMODEL);
    // h += gu @ Wd
    gemm_kernel<2><<<dim3(DMODEL / 64, TOK / 128), 256, 0, stream>>>(
        gub, WdT, nullptr, nullptr, hbuf, TOK, DMODEL, DFF);
    // router -> per-token scalar s;  out = s * colsum(Wout) + bout
    router_kernel<<<TOK, 256, 0, stream>>>(hbuf, Wr, br, sbuf);
    wsum_kernel<<<DOUT, 256, 0, stream>>>(Wout, wsum);
    final_kernel<<<TOK, DOUT, 0, stream>>>(sbuf, wsum, bout, out);
}
